// MHSA_67525475827795
// MI455X (gfx1250) — matrix operand order repaired and checked by reading, not compiled, not run on hardware
//
#include <hip/hip_runtime.h>

typedef _Float16 f16;
typedef __attribute__((ext_vector_type(16))) _Float16 v16h;
typedef __attribute__((ext_vector_type(8)))  _Float16 v8h;
typedef __attribute__((ext_vector_type(4)))  _Float16 v4h;
typedef __attribute__((ext_vector_type(8)))  float    v8f;

#define D_MODEL 1024
#define NH      16
#define HD      64
#define SEQ     2048
#define BATCH   2
#define M_ROWS  (BATCH * SEQ)     // 4096
#define N_QKV   (3 * D_MODEL)     // 3072

static __device__ __forceinline__ v8f wmma_f16(v16h a, v16h b, v8f c) {
  // (neg_a, A, neg_b, B, c_mod, C, reuse_a, reuse_b)
  return __builtin_amdgcn_wmma_f32_16x16x32_f16(false, a, false, b, (short)0, c,
                                                false, false);
}

static __device__ __forceinline__ v16h cat8(v8h lo, v8h hi) {
  return __builtin_shufflevector(lo, hi, 0, 1, 2, 3, 4, 5, 6, 7, 8, 9, 10, 11,
                                 12, 13, 14, 15);
}

// A-fragment load: row pointer + k offset; lane loads contiguous k [2*koff8, 2*koff8+16)
static __device__ __forceinline__ v16h lda(const f16* rowp, int kk, int koff8) {
  return cat8(*(const v8h*)(rowp + kk + 2 * koff8),
              *(const v8h*)(rowp + kk + 2 * koff8 + 8));
}

// ---------------------------------------------------------------------------
// f32 -> f16 convert, 4 elements per thread
// ---------------------------------------------------------------------------
__global__ __launch_bounds__(256) void cvt_f32_f16(const float* __restrict__ src,
                                                   f16* __restrict__ dst,
                                                   int n4) {
  int i = blockIdx.x * blockDim.x + threadIdx.x;
  if (i < n4) {
    float4 v = reinterpret_cast<const float4*>(src)[i];
    v4h h = {(f16)v.x, (f16)v.y, (f16)v.z, (f16)v.w};
    reinterpret_cast<v4h*>(dst)[i] = h;
  }
}

// ---------------------------------------------------------------------------
// QKV projection: C[m,n] = sum_k X[m,k] * Wqkv[n,k]
// Wave computes a 32x64 tile; block = 8 waves = 256x64. Software-pipelined:
// k-step (kk+32) fragments are loaded before the WMMAs of k-step kk.
// Output scattered into Q [b,h,s,hd], K [b,h,s,hd], Vt [b,h,hd,s] (f16).
// ---------------------------------------------------------------------------
__global__ __launch_bounds__(256) void qkv_gemm(const f16* __restrict__ X,
                                                const f16* __restrict__ W,
                                                f16* __restrict__ Q,
                                                f16* __restrict__ K,
                                                f16* __restrict__ Vt) {
  const int lane = threadIdx.x & 31;
  const int wid = threadIdx.x >> 5;
  const int lo = lane & 15, hi = lane >> 4;
  const int koff8 = hi * 8, koff16 = hi * 16;
  const int mbase = blockIdx.x * 256 + wid * 32;
  const int nbase = blockIdx.y * 64;

  v8f acc[2][4] = {};
  const f16* arow[2];
  arow[0] = X + (size_t)(mbase + lo) * D_MODEL;
  arow[1] = arow[0] + 16 * D_MODEL;
  const f16* brow[4];
#pragma unroll
  for (int t = 0; t < 4; ++t)
    brow[t] = W + (size_t)(nbase + t * 16 + lo) * D_MODEL + koff16;

  v16h a[2], b[4];
#pragma unroll
  for (int mi = 0; mi < 2; ++mi) a[mi] = lda(arow[mi], 0, koff8);
#pragma unroll
  for (int t = 0; t < 4; ++t) b[t] = *(const v16h*)(brow[t]);

  for (int kk = 0; kk < D_MODEL; kk += 32) {
    const int kn = (kk + 32) & (D_MODEL - 1);  // wrap keeps last prefetch legal
    v16h an[2], bn[4];
#pragma unroll
    for (int mi = 0; mi < 2; ++mi) an[mi] = lda(arow[mi], kn, koff8);
#pragma unroll
    for (int t = 0; t < 4; ++t) bn[t] = *(const v16h*)(brow[t] + kn);
#pragma unroll
    for (int mi = 0; mi < 2; ++mi)
#pragma unroll
      for (int t = 0; t < 4; ++t)
        acc[mi][t] = wmma_f16(a[mi], b[t], acc[mi][t]);
#pragma unroll
    for (int mi = 0; mi < 2; ++mi) a[mi] = an[mi];
#pragma unroll
    for (int t = 0; t < 4; ++t) b[t] = bn[t];
  }

#pragma unroll
  for (int t = 0; t < 4; ++t) {
    const int n = nbase + t * 16 + lo;
    const int which = n >> 10;  // 0=Q, 1=K, 2=V
    const int dmi = n & 1023;
    const int h = dmi >> 6, hd = dmi & 63;
#pragma unroll
    for (int mi = 0; mi < 2; ++mi) {
      const int m0 = mbase + mi * 16 + hi * 8;  // 8 consecutive rows
      const int bb = m0 >> 11, s0 = m0 & 2047;
      if (which == 2) {
        // Vt[b][h][hd][s]: 8 consecutive s are contiguous -> packed 16B store
        v8h pk;
#pragma unroll
        for (int r = 0; r < 8; ++r) pk[r] = (f16)acc[mi][t][r];
        *(v8h*)(Vt + ((size_t)(bb * NH + h) * HD + hd) * SEQ + s0) = pk;
      } else {
        f16* dst = (which == 0 ? Q : K) +
                   ((size_t)(bb * NH + h) * SEQ + s0) * HD + hd;
#pragma unroll
        for (int r = 0; r < 8; ++r) dst[(size_t)r * HD] = (f16)acc[mi][t][r];
      }
    }
  }
}

// ---------------------------------------------------------------------------
// Flash attention: per (b,h) and 64-row q block; 4 waves x 16 q rows.
// Online softmax over 32-key chunks. K fragments for the next chunk and V
// fragments for the current chunk are issued BEFORE the softmax VALU work so
// exp2/shuffles overlap the global loads. P re-layout via per-wave LDS bounce.
// ---------------------------------------------------------------------------
__global__ __launch_bounds__(128) void attn_kernel(const f16* __restrict__ Q,
                                                   const f16* __restrict__ K,
                                                   const f16* __restrict__ Vt,
                                                   f16* __restrict__ O) {
  __shared__ f16 p_lds[4][16][32];  // 4 KB: one 16x32 P tile per wave

  const int lane = threadIdx.x & 31;
  const int wid = threadIdx.x >> 5;
  const int lo = lane & 15, hi = lane >> 4;
  const int koff8 = hi * 8, koff16 = hi * 16;
  const int bh = blockIdx.y;  // b*NH + h
  const int qbase = blockIdx.x * 64 + wid * 16;

  const f16* qptr = Q + ((size_t)bh * SEQ + qbase) * HD;
  const f16* kptr = K + (size_t)bh * SEQ * HD;
  const f16* vptr = Vt + (size_t)bh * HD * SEQ;

  // Q fragments (A layout), hd chunks [0,32) and [32,64)
  v16h qf[2];
#pragma unroll
  for (int c = 0; c < 2; ++c) qf[c] = lda(qptr + lo * HD, c * 32, koff8);

  v8f oacc[4] = {};
  float mrow[8], lrow[8];
#pragma unroll
  for (int r = 0; r < 8; ++r) {
    mrow[r] = -__builtin_inff();
    lrow[r] = 0.0f;
  }

  const float sc = 0.125f * 1.44269504088896f;  // 1/sqrt(64) * log2(e)

  // preload K fragments for first key chunk: kf[key-tile t][hd-chunk c]
  v16h kf[2][2];
#pragma unroll
  for (int t = 0; t < 2; ++t) {
    const f16* krow = kptr + (size_t)(t * 16 + lo) * HD + koff16;
    kf[t][0] = *(const v16h*)(krow);
    kf[t][1] = *(const v16h*)(krow + 32);
  }

  for (int kb = 0; kb < SEQ; kb += 32) {
    // ---- scores: two 16-key tiles, K-dim = hd = 64 (two chained WMMAs) ----
    v8f s01[2];
#pragma unroll
    for (int t = 0; t < 2; ++t) {
      v8f cacc = {};
      cacc = wmma_f16(qf[0], kf[t][0], cacc);
      cacc = wmma_f16(qf[1], kf[t][1], cacc);
      s01[t] = cacc;
    }

    // ---- issue next-chunk K loads and current-chunk V loads early ----
    const int kbn = (kb + 32) & (SEQ - 1);  // wrap keeps last prefetch legal
    v16h kfn[2][2];
#pragma unroll
    for (int t = 0; t < 2; ++t) {
      const f16* krow = kptr + (size_t)(kbn + t * 16 + lo) * HD + koff16;
      kfn[t][0] = *(const v16h*)(krow);
      kfn[t][1] = *(const v16h*)(krow + 32);
    }
    v16h vf[4];
#pragma unroll
    for (int t = 0; t < 4; ++t)
      vf[t] = *(const v16h*)(vptr + (size_t)(t * 16 + lo) * SEQ + kb + koff16);

    // ---- online softmax (rows spread over 16-lane groups) ----
    float alpha[8];
#pragma unroll
    for (int r = 0; r < 8; ++r) {
      float a0 = s01[0][r] * sc;
      float a1 = s01[1][r] * sc;
      float mx = fmaxf(a0, a1);
#pragma unroll
      for (int off = 1; off < 16; off <<= 1)
        mx = fmaxf(mx, __shfl_xor(mx, off, 32));
      const float mnew = fmaxf(mrow[r], mx);
      const float al = __builtin_exp2f(mrow[r] - mnew);
      const float p0 = __builtin_exp2f(a0 - mnew);
      const float p1 = __builtin_exp2f(a1 - mnew);
      float rs = p0 + p1;
#pragma unroll
      for (int off = 1; off < 16; off <<= 1) rs += __shfl_xor(rs, off, 32);
      lrow[r] = lrow[r] * al + rs;
      mrow[r] = mnew;
      alpha[r] = al;
      // stash P (C layout -> LDS [row][col])
      p_lds[wid][hi * 8 + r][lo] = (f16)p0;
      p_lds[wid][hi * 8 + r][16 + lo] = (f16)p1;
    }

    // same-wave DS is in-order; this is compiler fence + explicit wait
    asm volatile("s_wait_dscnt 0x0" ::: "memory");

    // ---- re-read P as A fragment, rescale O, accumulate PV ----
    {
      const f16* pr = &p_lds[wid][lo][0];
      v16h pf = cat8(*(const v8h*)(pr + 2 * koff8), *(const v8h*)(pr + 2 * koff8 + 8));
#pragma unroll
      for (int t = 0; t < 4; ++t) {
#pragma unroll
        for (int r = 0; r < 8; ++r) oacc[t][r] *= alpha[r];
        oacc[t] = wmma_f16(pf, vf[t], oacc[t]);
      }
    }
    // WAR on p_lds across iterations: compiler fence (HW DS is in-order)
    asm volatile("" ::: "memory");

#pragma unroll
    for (int t = 0; t < 2; ++t) {
      kf[t][0] = kfn[t][0];
      kf[t][1] = kfn[t][1];
    }
  }

  // ---- epilogue: divide by l, store O as [b, s, d_model] f16 ----
  const int h = bh & (NH - 1);
  const int bi = bh >> 4;
#pragma unroll
  for (int t = 0; t < 4; ++t) {
#pragma unroll
    for (int r = 0; r < 8; ++r) {
      const int s = qbase + hi * 8 + r;
      const float v = oacc[t][r] / lrow[r];
      O[((size_t)bi * SEQ + s) * D_MODEL + h * HD + t * 16 + lo] = (f16)v;
    }
  }
}

// ---------------------------------------------------------------------------
// Output projection: out[m,n] = sum_k O[m,k]*Wout[n,k] + bias[n]  (fp32 out)
// Wave computes 32x64; software-pipelined like qkv_gemm.
// ---------------------------------------------------------------------------
__global__ __launch_bounds__(256) void out_gemm(const f16* __restrict__ A,
                                                const f16* __restrict__ W,
                                                const float* __restrict__ bias,
                                                float* __restrict__ out) {
  const int lane = threadIdx.x & 31;
  const int wid = threadIdx.x >> 5;
  const int lo = lane & 15, hi = lane >> 4;
  const int koff8 = hi * 8, koff16 = hi * 16;
  const int mbase = blockIdx.x * 256 + wid * 32;
  const int nbase = blockIdx.y * 64;

  v8f acc[2][4] = {};
  const f16* arow[2];
  arow[0] = A + (size_t)(mbase + lo) * D_MODEL;
  arow[1] = arow[0] + 16 * D_MODEL;
  const f16* brow[4];
#pragma unroll
  for (int t = 0; t < 4; ++t)
    brow[t] = W + (size_t)(nbase + t * 16 + lo) * D_MODEL + koff16;

  v16h a[2], b[4];
#pragma unroll
  for (int mi = 0; mi < 2; ++mi) a[mi] = lda(arow[mi], 0, koff8);
#pragma unroll
  for (int t = 0; t < 4; ++t) b[t] = *(const v16h*)(brow[t]);

  for (int kk = 0; kk < D_MODEL; kk += 32) {
    const int kn = (kk + 32) & (D_MODEL - 1);
    v16h an[2], bn[4];
#pragma unroll
    for (int mi = 0; mi < 2; ++mi) an[mi] = lda(arow[mi], kn, koff8);
#pragma unroll
    for (int t = 0; t < 4; ++t) bn[t] = *(const v16h*)(brow[t] + kn);
#pragma unroll
    for (int mi = 0; mi < 2; ++mi)
#pragma unroll
      for (int t = 0; t < 4; ++t)
        acc[mi][t] = wmma_f16(a[mi], b[t], acc[mi][t]);
#pragma unroll
    for (int mi = 0; mi < 2; ++mi) a[mi] = an[mi];
#pragma unroll
    for (int t = 0; t < 4; ++t) b[t] = bn[t];
  }

#pragma unroll
  for (int t = 0; t < 4; ++t) {
    const int n = nbase + t * 16 + lo;
    const float bn2 = bias[n];
#pragma unroll
    for (int mi = 0; mi < 2; ++mi) {
#pragma unroll
      for (int r = 0; r < 8; ++r) {
        const int m = mbase + mi * 16 + hi * 8 + r;
        out[(size_t)m * D_MODEL + n] = acc[mi][t][r] + bn2;
      }
    }
  }
}

// ---------------------------------------------------------------------------
// Host launcher
// ---------------------------------------------------------------------------
extern "C" void kernel_launch(void* const* d_in, const int* in_sizes, int n_in,
                              void* d_out, int out_size, void* d_ws,
                              size_t ws_size, hipStream_t stream) {
  const float* x = (const float*)d_in[0];      // [2,2048,1024]
  const float* W_qkv = (const float*)d_in[1];  // [3072,1024]
  const float* W_out = (const float*)d_in[2];  // [1024,1024]
  const float* b_out = (const float*)d_in[3];  // [1024]
  float* out = (float*)d_out;

  char* ws = (char*)d_ws;
  f16* xh = (f16*)(ws);                              //  8 MiB
  f16* wqh = (f16*)(ws + ((size_t)8 << 20));         //  6 MiB
  f16* woh = (f16*)(ws + ((size_t)14 << 20));        //  2 MiB
  f16* Qb = (f16*)(ws + ((size_t)16 << 20));         //  8 MiB
  f16* Kb = (f16*)(ws + ((size_t)24 << 20));         //  8 MiB
  f16* Vtb = (f16*)(ws + ((size_t)32 << 20));        //  8 MiB
  f16* Ob = (f16*)(ws + ((size_t)40 << 20));         //  8 MiB  (48 MiB total)

  const int n1 = M_ROWS * D_MODEL / 4;
  const int n2 = N_QKV * D_MODEL / 4;
  const int n3 = D_MODEL * D_MODEL / 4;
  cvt_f32_f16<<<(n1 + 255) / 256, 256, 0, stream>>>(x, xh, n1);
  cvt_f32_f16<<<(n2 + 255) / 256, 256, 0, stream>>>(W_qkv, wqh, n2);
  cvt_f32_f16<<<(n3 + 255) / 256, 256, 0, stream>>>(W_out, woh, n3);

  qkv_gemm<<<dim3(M_ROWS / 256, N_QKV / 64), 256, 0, stream>>>(xh, wqh, Qb, Kb,
                                                               Vtb);
  attn_kernel<<<dim3(SEQ / 64, BATCH * NH), 128, 0, stream>>>(Qb, Kb, Vtb, Ob);
  out_gemm<<<dim3(M_ROWS / 256, D_MODEL / 64), 256, 0, stream>>>(Ob, woh, b_out,
                                                                 out);
}
